// AttentionAggregator_8005819040417
// MI455X (gfx1250) — matrix operand order repaired and checked by reading, not compiled, not run on hardware
//
#include <hip/hip_runtime.h>
#include <hip/hip_fp16.h>

// CDNA5 (gfx1250) fused attention aggregator.
// One wave32 per node; m = neigh@W2 via v_wmma_f32_16x16x32_f16.

typedef __attribute__((ext_vector_type(16))) _Float16 v16h;
typedef __attribute__((ext_vector_type(8)))  _Float16 v8h;
typedef __attribute__((ext_vector_type(8)))  float    v8f;

#define KNEI 32
#define DIN  128
#define DOUT 128
#define NPB  6                    // nodes (waves) per block
#define NTHREADS (NPB * 32)
#define AST  136                  // LDS f16 row stride (272 B: 16B-aligned, bank-conflict-free)

// Setup: W2 (f32 row-major [d][e]) -> W2T (f16 row-major [e][d]) in workspace.
__global__ void w2t_f16_kernel(const float* __restrict__ W2, _Float16* __restrict__ W2T) {
  int idx = blockIdx.x * 256 + threadIdx.x;   // 0..16383
  int d = idx >> 7;
  int e = idx & 127;
  W2T[e * DIN + d] = (_Float16)W2[idx];
}

__global__ __launch_bounds__(NTHREADS) void attn_agg_kernel(
    const float* __restrict__ self_vecs,      // [N,128]
    const float* __restrict__ neigh,          // [N,32,128]
    const float* __restrict__ ndt,            // [N,32]
    const float* __restrict__ W1,             // [128,128]
    const float* __restrict__ W3,             // [1,128]
    const float* __restrict__ W4,             // [128,1]
    const float* __restrict__ b4,             // [1]
    const _Float16* __restrict__ W2T,         // [128,128] f16, transposed W2
    float* __restrict__ out,                  // [N,128]
    float* __restrict__ score_out,            // [N,32]
    int N)
{
  __shared__ __attribute__((aligned(16))) _Float16 nf16[NPB][KNEI * AST];
  __shared__ float w3s[DIN];
  __shared__ float w4s[DOUT];
  __shared__ float ps[NPB][KNEI];             // m.W4 partials, then softmax probs

  const int tid  = threadIdx.x;
  const int wv   = tid >> 5;
  const int lane = tid & 31;
  const int l15  = lane & 15;
  const int hh   = lane >> 4;
  const int node = blockIdx.x * NPB + wv;
  const bool active = node < N;

  // ---- Phase 0: stage this node's 32x128 tile in LDS as f16 --------------
  if (active) {
    const float* nb = neigh + (size_t)node * (KNEI * DIN);
#pragma unroll 4
    for (int k = 0; k < KNEI; ++k) {
      float4 v = ((const float4*)(nb + (size_t)k * DIN))[lane];
      union { _Float16 h[4]; unsigned long long u; } pk;
      pk.h[0] = (_Float16)v.x; pk.h[1] = (_Float16)v.y;
      pk.h[2] = (_Float16)v.z; pk.h[3] = (_Float16)v.w;
      *(unsigned long long*)&nf16[wv][k * AST + 4 * lane] = pk.u;
    }
  }
  if (tid < DIN) { w3s[tid] = W3[tid]; w4s[tid] = W4[tid]; }
  __syncthreads();

  float qs = 0.f, tsv = 0.f;
  if (active) {
    // ---- q = self @ W1 (fp32); downstream only needs qs = q.W4 -----------
    const float* sv = self_vecs + (size_t)node * DIN;
    float q0 = 0.f, q1 = 0.f, q2 = 0.f, q3 = 0.f;
    for (int d = 0; d < DIN; ++d) {
      float s = sv[d];
      const float* wr = W1 + d * DOUT;
      q0 = fmaf(s, wr[lane],      q0);
      q1 = fmaf(s, wr[lane + 32], q1);
      q2 = fmaf(s, wr[lane + 64], q2);
      q3 = fmaf(s, wr[lane + 96], q3);
    }
    qs = q0 * w4s[lane] + q1 * w4s[lane + 32] + q2 * w4s[lane + 64] + q3 * w4s[lane + 96];
    qs += __shfl_xor(qs, 16, 32);
    qs += __shfl_xor(qs,  8, 32);
    qs += __shfl_xor(qs,  4, 32);
    qs += __shfl_xor(qs,  2, 32);
    qs += __shfl_xor(qs,  1, 32);

    // ---- t path: ts[k] = sum_e sigmoid(dt_k * W3[e]) * W4[e], lane == k --
    float dtv = ndt[(size_t)node * KNEI + lane];
    for (int e = 0; e < DIN; ++e) {
      float x  = dtv * w3s[e];
      float sg = 1.0f / (1.0f + __expf(-x));
      tsv = fmaf(sg, w4s[e], tsv);
    }

    // ---- m = neigh @ W2 via WMMA, reduced in-register against W4 ---------
    // A frag (16x32 f16): lane row = l15, 16 halves at K = 32kb + {8hh..8hh+7, 16+8hh..+7}
    // B frag (32x16 f16): lane col = l15, 16 halves at K = 32kb + {8hh..8hh+7, 16+8hh..+7}
    // C/D (16x16 f32):   lane col = l15, VGPR r -> row r + 8hh
#pragma unroll
    for (int mt = 0; mt < 2; ++mt) {
      v8f acc[8] = {};
#pragma unroll
      for (int kb = 0; kb < 4; ++kb) {
        const _Float16* ar = &nf16[wv][(16 * mt + l15) * AST + 32 * kb + 8 * hh];
        v8h alo = *(const v8h*)ar;
        v8h ahi = *(const v8h*)(ar + 16);
        v16h A;
#pragma unroll
        for (int i = 0; i < 8; ++i) { A[i] = alo[i]; A[i + 8] = ahi[i]; }
#pragma unroll
        for (int nt = 0; nt < 8; ++nt) {
          const _Float16* br = W2T + (16 * nt + l15) * DIN + 32 * kb + 8 * hh;
          v8h blo = *(const v8h*)br;
          v8h bhi = *(const v8h*)(br + 16);
          v16h B;
#pragma unroll
          for (int i = 0; i < 8; ++i) { B[i] = blo[i]; B[i + 8] = bhi[i]; }
          acc[nt] = __builtin_amdgcn_wmma_f32_16x16x32_f16(
              false, A, false, B, (short)0, acc[nt], false, false);
        }
      }
      // acc[nt][r] = m[16*mt + r + 8*hh][16*nt + l15]; dot with W4 + 16-lane sum
#pragma unroll
      for (int r = 0; r < 8; ++r) {
        float part = 0.f;
#pragma unroll
        for (int nt = 0; nt < 8; ++nt)
          part = fmaf(acc[nt][r], w4s[16 * nt + l15], part);
        part += __shfl_xor(part, 1, 32);
        part += __shfl_xor(part, 2, 32);
        part += __shfl_xor(part, 4, 32);
        part += __shfl_xor(part, 8, 32);
        if (l15 == r) ps[wv][16 * mt + 8 * hh + r] = part;   // = m[k].W4
      }
    }
  }
  __syncthreads();

  // ---- score = relu(q.W4 + m.W4 + t.W4 + b4); softmax over K (lane == k) -
  if (active) {
    float lin = ps[wv][lane] + qs + tsv + b4[0];
    float s = fmaxf(lin, 0.f);
    float mx = s;
    mx = fmaxf(mx, __shfl_xor(mx, 16, 32));
    mx = fmaxf(mx, __shfl_xor(mx,  8, 32));
    mx = fmaxf(mx, __shfl_xor(mx,  4, 32));
    mx = fmaxf(mx, __shfl_xor(mx,  2, 32));
    mx = fmaxf(mx, __shfl_xor(mx,  1, 32));
    float ex = __expf(s - mx);
    float sum = ex;
    sum += __shfl_xor(sum, 16, 32);
    sum += __shfl_xor(sum,  8, 32);
    sum += __shfl_xor(sum,  4, 32);
    sum += __shfl_xor(sum,  2, 32);
    sum += __shfl_xor(sum,  1, 32);
    float p = ex / sum;
    score_out[(size_t)node * KNEI + lane] = p;
    ps[wv][lane] = p;                         // own-slot read->write, no hazard
  }
  __syncthreads();

  // ---- output[n] = sum_k p[k] * neigh[n][k]  (fp32, L2-hot re-read) ------
  if (active) {
    const float* nb = neigh + (size_t)node * (KNEI * DIN);
    float ax = 0.f, ay = 0.f, az = 0.f, aw = 0.f;
#pragma unroll 4
    for (int k = 0; k < KNEI; ++k) {
      float pk = ps[wv][k];
      float4 v = ((const float4*)(nb + (size_t)k * DIN))[lane];
      ax = fmaf(pk, v.x, ax); ay = fmaf(pk, v.y, ay);
      az = fmaf(pk, v.z, az); aw = fmaf(pk, v.w, aw);
    }
    float4 o; o.x = ax; o.y = ay; o.z = az; o.w = aw;
    ((float4*)(out + (size_t)node * DIN))[lane] = o;
  }
}

extern "C" void kernel_launch(void* const* d_in, const int* in_sizes, int n_in,
                              void* d_out, int out_size, void* d_ws, size_t ws_size,
                              hipStream_t stream) {
  const float* self_vecs = (const float*)d_in[0];
  const float* neigh     = (const float*)d_in[1];
  const float* ndt       = (const float*)d_in[2];
  const float* W1        = (const float*)d_in[3];
  const float* W2        = (const float*)d_in[4];
  const float* W3        = (const float*)d_in[5];
  const float* W4        = (const float*)d_in[6];
  const float* b4        = (const float*)d_in[7];

  const int N = in_sizes[0] / DIN;            // 50000

  float* out   = (float*)d_out;               // [N,128]
  float* score = out + (size_t)N * DIN;       // [N,32] (concatenated outputs)
  _Float16* W2T = (_Float16*)d_ws;            // 32 KB workspace

  w2t_f16_kernel<<<64, 256, 0, stream>>>(W2, W2T);

  int blocks = (N + NPB - 1) / NPB;
  attn_agg_kernel<<<blocks, NTHREADS, 0, stream>>>(
      self_vecs, neigh, ndt, W1, W3, W4, b4, W2T, out, score, N);
}
